// LSTMDecode_69861938037069
// MI455X (gfx1250) — compile-verified, matrix operand order checked by reading, not run on hardware
//
#include <hip/hip_runtime.h>
#include <hip/hip_bf16.h>
#include <math.h>

// Problem constants
#define Bx   4096
#define Tt   96
#define INx  64
#define Hx   512
#define G3   1536          // 3*H
#define KTOT 608           // 64 (x) + 1 (y_last) + 31 (pad) + 512 (h)
#define NCH  19            // KTOT / 32
#define XCH  3             // first 3 K-chunks belong to the X part (i_n), rest to h (h_n)
#define SROW 616           // LDS row stride in bf16 elements (608 + 8 pad)

typedef __attribute__((ext_vector_type(16))) __bf16 v16bf;
typedef __attribute__((ext_vector_type(8)))  __bf16 v8bf;
typedef __attribute__((ext_vector_type(8)))  float  v8f;

// ---------------------------------------------------------------------------
// Pack [W_ih | 0 | W_hh] (g-major source) into a WMMA-B fragment-native bf16
// layout: frag index ((kc*96 + gt)*32 + lane)*16 + j, so each lane's 16
// operand values for one 32x16 B tile are 32 contiguous bytes.
// ---------------------------------------------------------------------------
__global__ __launch_bounds__(256) void prep_wfrag(const float* __restrict__ W_ih,
                                                  const float* __restrict__ W_hh,
                                                  __bf16* __restrict__ wfrag) {
  int f = blockIdx.x * 256 + threadIdx.x;
  if (f >= KTOT * G3) return;
  int j    = f & 15;
  int lane = (f >> 4) & 31;
  int rest = f >> 9;              // kc*96 + gt
  int gt   = rest % 96;
  int kc   = rest / 96;
  // bf16 A/B operand layout: lanes 0-15 hold K {0..7,16..23}, lanes 16-31 K {8..15,24..31}
  int klocal = (lane < 16) ? ((j < 8) ? j : j + 8)
                           : ((j < 8) ? j + 8 : j + 16);
  int k = kc * 32 + klocal;
  int g = gt * 16 + (lane & 15);
  float v = 0.0f;
  if (k < INx + 1)      v = W_ih[g * (INx + 1) + k];   // x part + y_last column (k==64)
  else if (k >= 96)     v = W_hh[g * Hx + (k - 96)];   // h part
  wfrag[f] = (__bf16)v;
}

// h0 = ht ; y0 = y_past[:, -1, 0]
__global__ __launch_bounds__(256) void init_state(const float* __restrict__ ht,
                                                  const float* __restrict__ y_past,
                                                  float* __restrict__ h0,
                                                  float* __restrict__ y0) {
  int i = blockIdx.x * 256 + threadIdx.x;
  if (i < Bx * Hx) h0[i] = ht[i];
  if (i < Bx)      y0[i] = y_past[i * 32 + 31];
}

// ---------------------------------------------------------------------------
// One GRU step. Block = 256 threads = 8 wave32s, computes a 32(batch) x
// 64(hcol) tile of h_next. Wave (wm,wn) owns the 16x16 subtile
// rows [rt*32 + wm*16, +16), cols [ct*64 + wn*16, +16).
// ---------------------------------------------------------------------------
__global__ __launch_bounds__(256) void gru_step(const float* __restrict__ x_fut,
                                                const float* __restrict__ b_ih,
                                                const float* __restrict__ b_hh,
                                                const __bf16* __restrict__ wfrag,
                                                const float* __restrict__ h_cur,
                                                const float* __restrict__ y_cur,
                                                float* __restrict__ h_next,
                                                int t) {
  __shared__ __align__(16) __bf16 sA[32 * SROW];

  const int ct   = blockIdx.x;          // 0..7  : 64-wide column tile of H
  const int rt   = blockIdx.y;          // 0..127: 32-high batch tile
  const int tid  = threadIdx.x;
  const int lane = tid & 31;
  const int wave = tid >> 5;
  const int wm   = wave >> 2;           // 0..1
  const int wn   = wave & 3;            // 0..3
  const int rowBase = rt * 32;

  // ---- stage augmented input rows [x_t | y_last | 0pad | h] as bf16 in LDS
  for (int s = tid; s < 32 * KTOT; s += 256) {
    int row = s / KTOT;
    int k   = s - row * KTOT;
    int b   = rowBase + row;
    float v;
    if (k < INx)          v = x_fut[(b * Tt + t) * INx + k];
    else if (k == INx)    v = y_cur[b];
    else if (k < 96)      v = 0.0f;
    else                  v = h_cur[b * Hx + (k - 96)];
    sA[row * SROW + k] = (__bf16)v;
  }
  __syncthreads();

  v8f accR = {}, accZ = {}, accIN = {}, accHN = {};

  const int gCol0 = ct * 64 + wn * 16;        // h column base (0..511)
  const int aRow  = wm * 16 + (lane & 15);    // A-fragment row within LDS tile
  const int kSel  = (lane >> 4) * 8;          // K phase per half-wave
  const int gt_r  = gCol0 >> 4;               // column-tile indices in wfrag
  const int gt_z  = (gCol0 + Hx)  >> 4;
  const int gt_n  = (gCol0 + 2*Hx) >> 4;

  for (int kc = 0; kc < NCH; ++kc) {
    const int k0 = kc * 32 + kSel;
    union { v16bf v; v8bf h[2]; } a;
    a.h[0] = *(const v8bf*)&sA[aRow * SROW + k0];
    a.h[1] = *(const v8bf*)&sA[aRow * SROW + k0 + 16];

    const v16bf br = *(const v16bf*)&wfrag[(((size_t)kc * 96 + gt_r) * 32 + lane) * 16];
    const v16bf bz = *(const v16bf*)&wfrag[(((size_t)kc * 96 + gt_z) * 32 + lane) * 16];
    const v16bf bn = *(const v16bf*)&wfrag[(((size_t)kc * 96 + gt_n) * 32 + lane) * 16];

    if (kc + 1 < NCH) {  // weights are L2-resident; keep the next chunk warm
      __builtin_prefetch(&wfrag[(((size_t)(kc + 1) * 96 + gt_r) * 32 + lane) * 16], 0, 3);
    }

    accR = __builtin_amdgcn_wmma_f32_16x16x32_bf16(false, a.v, false, br,
                                                   (short)0, accR, false, false);
    accZ = __builtin_amdgcn_wmma_f32_16x16x32_bf16(false, a.v, false, bz,
                                                   (short)0, accZ, false, false);
    if (kc < XCH)
      accIN = __builtin_amdgcn_wmma_f32_16x16x32_bf16(false, a.v, false, bn,
                                                      (short)0, accIN, false, false);
    else
      accHN = __builtin_amdgcn_wmma_f32_16x16x32_bf16(false, a.v, false, bn,
                                                      (short)0, accHN, false, false);
  }

  // ---- gate math + h update (C/D layout: N = lane&15, M = vgpr + 8*(lane>=16))
  const int n  = lane & 15;
  const int g  = gCol0 + n;
  const float brz = b_ih[g]          + b_hh[g];
  const float bzz = b_ih[Hx + g]     + b_hh[Hx + g];
  const float bin = b_ih[2 * Hx + g];
  const float bhn = b_hh[2 * Hx + g];
  const int   mO  = (lane >> 4) * 8;

#pragma unroll
  for (int i = 0; i < 8; ++i) {
    int b = rowBase + wm * 16 + mO + i;
    float r  = 1.0f / (1.0f + __expf(-(accR[i] + brz)));
    float z  = 1.0f / (1.0f + __expf(-(accZ[i] + bzz)));
    float nn = tanhf(accIN[i] + bin + r * (accHN[i] + bhn));
    float ho = h_cur[b * Hx + g];
    h_next[b * Hx + g] = (1.0f - z) * nn + z * ho;
  }
}

// ---------------------------------------------------------------------------
// y = h_next @ W_lin^T + b_lin ; one wave per batch row, shuffle reduction.
// Writes y ping-pong buffer (next step's y_last) and the output column t.
// ---------------------------------------------------------------------------
__global__ __launch_bounds__(256) void gru_out(const float* __restrict__ h_next,
                                               const float* __restrict__ W_lin,
                                               const float* __restrict__ b_lin,
                                               float* __restrict__ y_next,
                                               float* __restrict__ out,
                                               int t) {
  const int lane = threadIdx.x & 31;
  const int wave = threadIdx.x >> 5;
  const int b = blockIdx.x * 8 + wave;
  float acc = 0.0f;
#pragma unroll
  for (int g = lane; g < Hx; g += 32) acc += h_next[b * Hx + g] * W_lin[g];
#pragma unroll
  for (int off = 16; off; off >>= 1) acc += __shfl_xor(acc, off, 32);
  if (lane == 0) {
    float y = acc + b_lin[0];
    y_next[b] = y;
    out[b * Tt + t] = y;
  }
}

extern "C" void kernel_launch(void* const* d_in, const int* in_sizes, int n_in,
                              void* d_out, int out_size, void* d_ws, size_t ws_size,
                              hipStream_t stream) {
  const float* x_fut  = (const float*)d_in[0];
  const float* y_past = (const float*)d_in[1];
  const float* ht     = (const float*)d_in[2];
  const float* W_ih   = (const float*)d_in[3];
  const float* W_hh   = (const float*)d_in[4];
  const float* b_ih   = (const float*)d_in[5];
  const float* b_hh   = (const float*)d_in[6];
  const float* W_lin  = (const float*)d_in[7];
  const float* b_lin  = (const float*)d_in[8];
  float* out = (float*)d_out;

  // workspace layout: [wfrag 2MB][h ping 8MB][h pong 8MB][y ping 16KB][y pong 16KB]
  char* ws = (char*)d_ws;
  __bf16* wfrag = (__bf16*)ws;
  size_t off = 2u << 20;
  float* hb0 = (float*)(ws + off);
  float* hb1 = (float*)(ws + off + (size_t)Bx * Hx * sizeof(float));
  off += (size_t)2 * Bx * Hx * sizeof(float);
  float* yb0 = (float*)(ws + off);
  float* yb1 = (float*)(ws + off + (size_t)Bx * sizeof(float));
  float* hbuf[2] = {hb0, hb1};
  float* ybuf[2] = {yb0, yb1};

  prep_wfrag<<<(KTOT * G3 + 255) / 256, 256, 0, stream>>>(W_ih, W_hh, wfrag);
  init_state<<<(Bx * Hx + 255) / 256, 256, 0, stream>>>(ht, y_past, hbuf[0], ybuf[0]);

  for (int t = 0; t < Tt; ++t) {
    const float* hc = hbuf[t & 1];
    float*       hn = hbuf[(t + 1) & 1];
    const float* yc = ybuf[t & 1];
    float*       yn = ybuf[(t + 1) & 1];
    dim3 grid(Hx / 64, Bx / 32);   // 8 x 128 workgroups
    gru_step<<<grid, 256, 0, stream>>>(x_fut, b_ih, b_hh, wfrag, hc, yc, hn, t);
    gru_out<<<Bx / 8, 256, 0, stream>>>(hn, W_lin, b_lin, yn, out, t);
  }
}
